// MultiHeadAttention_48258252538318
// MI455X (gfx1250) — matrix operand order repaired and checked by reading, not compiled, not run on hardware
//
#include <hip/hip_runtime.h>
#include <hip/hip_bf16.h>

typedef __attribute__((ext_vector_type(16))) __bf16 bf16x16;
typedef __attribute__((ext_vector_type(8)))  float  f32x8;

#define WMMA_BF16(a, b, c) \
  __builtin_amdgcn_wmma_f32_16x16x32_bf16(false, (a), false, (b), (short)0, (c), false, false)

// Async global->LDS 16-byte copy (CDNA5, ASYNCcnt-tracked).
__device__ __forceinline__ void async_copy_b128(unsigned lds_off, const void* gptr) {
  asm volatile("global_load_async_to_lds_b128 %0, %1, off"
               :: "v"(lds_off), "v"((unsigned long long)(size_t)gptr)
               : "memory");
}
__device__ __forceinline__ void wait_asynccnt0() {
  asm volatile("s_wait_asynccnt 0x0" ::: "memory");
}

// ds_swizzle xor-reduction step (group-of-32 encoding: [14:10]=xor, [4:0]=and).
template <int XOR>
__device__ __forceinline__ float swz_max_f32(float v) {
  constexpr int imm = (XOR << 10) | 0x1f;
  const int s = __builtin_amdgcn_ds_swizzle(__builtin_bit_cast(int, v), imm);
  return __builtin_fmaxf(v, __builtin_bit_cast(float, s));
}

__device__ __forceinline__ unsigned pack_bf16(__bf16 lo, __bf16 hi) {
  const unsigned a = __builtin_bit_cast(unsigned short, lo);
  const unsigned b = __builtin_bit_cast(unsigned short, hi);
  return a | (b << 16);
}

// ---------------------------------------------------------------------------
// GEMM: Y[M,N] = A[M,K] @ W[K,N] + bias[N]
// A fp32 or bf16 (converted to bf16 in LDS staging); W,bias fp32.
// Block tile 128x128, 256 threads = 8 wave32 (4x2), wave tile 32x64.
// W tile stored transposed in LDS; all LDS stores are packed b32.
// ---------------------------------------------------------------------------
template <typename TA, bool OUT_BF16>
__global__ void gemm_bf16_wmma(const TA* __restrict__ A, const float* __restrict__ W,
                               const float* __restrict__ bias, void* __restrict__ outp,
                               int M, int N, int K) {
  __shared__ unsigned As32[128 * 16];   // [m][k] as packed bf16 pairs
  __shared__ unsigned Bt32[128 * 16];   // [n][k] transposed W tile, packed
  const __bf16* As = (const __bf16*)As32;
  const __bf16* Bt = (const __bf16*)Bt32;

  const int tid  = threadIdx.x;
  const int wave = tid >> 5;
  const int lane = tid & 31;
  const int half = lane >> 4;
  const int lr   = lane & 15;
  const int wm   = wave >> 1;   // 0..3 (rows)
  const int wn   = wave & 1;    // 0..1 (cols)
  const int bm   = blockIdx.x * 128;
  const int bn   = blockIdx.y * 128;

  f32x8 acc[2][4];
#pragma unroll
  for (int ms = 0; ms < 2; ++ms)
#pragma unroll
    for (int ns = 0; ns < 4; ++ns)
      acc[ms][ns] = (f32x8){0.f, 0.f, 0.f, 0.f, 0.f, 0.f, 0.f, 0.f};

  for (int k0 = 0; k0 < K; k0 += 32) {
    // ---- stage A tile (128x32) into LDS, packed b32 stores ----
    {
      const int r = tid >> 3;        // 0..31
      const int c = (tid & 7) * 4;   // 0..28
#pragma unroll
      for (int i = 0; i < 4; ++i) {
        const int row = r + 32 * i;
        const TA* src = A + (size_t)(bm + row) * K + k0 + c;
        const __bf16 v0 = (__bf16)(float)src[0], v1 = (__bf16)(float)src[1];
        const __bf16 v2 = (__bf16)(float)src[2], v3 = (__bf16)(float)src[3];
        As32[row * 16 + (c >> 1)]     = pack_bf16(v0, v1);
        As32[row * 16 + (c >> 1) + 1] = pack_bf16(v2, v3);
      }
    }
    // ---- stage W tile (32x128) transposed; 2k x 2n per thread, packed ----
    {
      const int nb = (tid & 63) * 2;   // 0..126
      const int kb = (tid >> 6) * 2;   // 0,2,4,6
#pragma unroll
      for (int i = 0; i < 4; ++i) {
        const int k = kb + 8 * i;      // 0..31 (pairs)
        const float* s0 = W + (size_t)(k0 + k) * N + bn + nb;
        const float* s1 = s0 + N;
        Bt32[(nb)     * 16 + (k >> 1)] = pack_bf16((__bf16)s0[0], (__bf16)s1[0]);
        Bt32[(nb + 1) * 16 + (k >> 1)] = pack_bf16((__bf16)s0[1], (__bf16)s1[1]);
      }
    }
    __syncthreads();

    // ---- A fragments: 16x32 (lane half selects K sub-range) ----
    bf16x16 afrag[2];
#pragma unroll
    for (int ms = 0; ms < 2; ++ms) {
      const __bf16* arow = &As[(32 * wm + 16 * ms + lr) * 32];
      bf16x16 f;
#pragma unroll
      for (int e = 0; e < 8; ++e) f[e] = arow[16 * half + e];
#pragma unroll
      for (int e = 0; e < 8; ++e) f[8 + e] = arow[16 * half + 8 + e];
      afrag[ms] = f;
    }
    // ---- B fragments: 32x16, contiguous per lane thanks to transpose ----
#pragma unroll
    for (int ns = 0; ns < 4; ++ns) {
      const __bf16* brow = &Bt[(64 * wn + 16 * ns + lr) * 32 + 16 * half];
      bf16x16 bfrag;
#pragma unroll
      for (int e = 0; e < 16; ++e) bfrag[e] = brow[e];
#pragma unroll
      for (int ms = 0; ms < 2; ++ms)
        acc[ms][ns] = WMMA_BF16(afrag[ms], bfrag, acc[ms][ns]);
    }
    __syncthreads();
  }

  // ---- epilogue: bias add, store (C layout: lane=N, M in VGPR index) ----
#pragma unroll
  for (int ns = 0; ns < 4; ++ns) {
    const int col = bn + 64 * wn + 16 * ns + lr;
    const float bv = bias[col];
#pragma unroll
    for (int ms = 0; ms < 2; ++ms)
#pragma unroll
      for (int r = 0; r < 8; ++r) {
        const int row = bm + 32 * wm + 16 * ms + r + 8 * half;
        const float v = acc[ms][ns][r] + bv;
        if constexpr (OUT_BF16)
          ((__bf16*)outp)[(size_t)row * N + col] = (__bf16)v;
        else
          ((float*)outp)[(size_t)row * N + col] = v;
      }
  }
}

// ---------------------------------------------------------------------------
// Flash attention over bf16 Q/K/V in [B,S,H*HD] layout; output bf16 same layout.
// Block = 4 waves x 16 query rows = 64 queries; streams 64-key blocks.
// K block: async global->LDS copy. V block: packed transposed LDS staging.
// Softmax denominator accumulated on the WMMA pipe via an all-ones B matrix.
// ---------------------------------------------------------------------------
__global__ void attn_kernel(const __bf16* __restrict__ Qb, const __bf16* __restrict__ Kb,
                            const __bf16* __restrict__ Vb, __bf16* __restrict__ Ob) {
  constexpr int Sq = 4096, Dm = 512, HDc = 64;
  const int qblk = blockIdx.x;
  const int h    = blockIdx.y;
  const int b    = blockIdx.z;
  const int tid  = threadIdx.x;
  const int wave = tid >> 5;
  const int lane = tid & 31;
  const int half = lane >> 4;
  const int lr   = lane & 15;

  const int    q0   = qblk * 64 + wave * 16;
  const size_t base = ((size_t)b * Sq) * Dm + (size_t)h * HDc;
  const __bf16* Qp = Qb + base;
  const __bf16* Kp = Kb + base;
  const __bf16* Vp = Vb + base;

  __shared__ __bf16   Ks[64 * 64];      // [key][hd]  (async-copied)
  __shared__ unsigned Vt32[64 * 32];    // [hd][key]  packed bf16 key-pairs
  __shared__ __bf16   Ps[4][16 * 64];   // wave-private P tiles (C->A layout)
  const __bf16* Vt  = (const __bf16*)Vt32;
  __bf16*       myP = &Ps[wave][0];
  const unsigned ks_base = (unsigned)(size_t)(void*)Ks;  // LDS byte offset

  // Q fragments resident for the whole pass (contraction HD=64 -> 2 k-steps)
  bf16x16 qfrag[2];
#pragma unroll
  for (int kk = 0; kk < 2; ++kk) {
    const __bf16* row = Qp + (size_t)(q0 + lr) * Dm + kk * 32;
    bf16x16 f;
#pragma unroll
    for (int e = 0; e < 8; ++e) f[e] = row[16 * half + e];
#pragma unroll
    for (int e = 0; e < 8; ++e) f[8 + e] = row[16 * half + 8 + e];
    qfrag[kk] = f;
  }

  bf16x16 onesfrag;   // B matrix of ones: L += P x ones gives row sums of P
#pragma unroll
  for (int e = 0; e < 16; ++e) onesfrag[e] = (__bf16)1.0f;

  float m_run[8];
  f32x8 Oacc[4], Lacc;
#pragma unroll
  for (int r = 0; r < 8; ++r) m_run[r] = -__builtin_huge_valf();
#pragma unroll
  for (int t = 0; t < 4; ++t) Oacc[t] = (f32x8){0.f, 0.f, 0.f, 0.f, 0.f, 0.f, 0.f, 0.f};
  Lacc = (f32x8){0.f, 0.f, 0.f, 0.f, 0.f, 0.f, 0.f, 0.f};

  const float inv_scale = 0.125f;  // 1/sqrt(HD)

  for (int k0 = 0; k0 < Sq; k0 += 64) {
    // ---- stage K block 64x64 via async global->LDS (ASYNCcnt path) ----
    {
      const int row = tid >> 3;      // 0..15
      const int seg = tid & 7;       // 8 x 16B per row
#pragma unroll
      for (int i = 0; i < 4; ++i) {
        const int r = row + 16 * i;
        const unsigned lds_off = ks_base + (unsigned)((r * 64 + seg * 8) * 2);
        async_copy_b128(lds_off, Kp + (size_t)(k0 + r) * Dm + seg * 8);
      }
    }
    // ---- stage V block transposed: Vt[hd][key], packed b32 (key pairs) ----
    {
      const int pr  = tid >> 3;      // key pair 0..15
      const int seg = tid & 7;       // 8 hd per segment
#pragma unroll
      for (int i = 0; i < 2; ++i) {
        const int key = 2 * pr + 32 * i;
        const __bf16* g0 = Vp + (size_t)(k0 + key) * Dm + seg * 8;
        const __bf16* g1 = g0 + Dm;
#pragma unroll
        for (int j = 0; j < 8; ++j)
          Vt32[(seg * 8 + j) * 32 + pr + 16 * i] = pack_bf16(g0[j], g1[j]);
      }
    }
    if (k0 + 64 < Sq)   // pull next K block toward caches (global_prefetch_b8)
      __builtin_prefetch(Kp + (size_t)(k0 + 64 + (tid >> 1)) * Dm, 0, 0);
    wait_asynccnt0();
    __syncthreads();

    // ---- S = Q K^T : 16 queries x 64 keys, contraction over HD ----
    f32x8 Sacc[4];
#pragma unroll
    for (int t = 0; t < 4; ++t) {
      f32x8 c = (f32x8){0.f, 0.f, 0.f, 0.f, 0.f, 0.f, 0.f, 0.f};
#pragma unroll
      for (int kk = 0; kk < 2; ++kk) {
        const __bf16* krow = &Ks[(16 * t + lr) * 64 + kk * 32 + 16 * half];
        bf16x16 bf;
#pragma unroll
        for (int e = 0; e < 16; ++e) bf[e] = krow[e];
        c = WMMA_BF16(qfrag[kk], bf, c);
      }
      Sacc[t] = c;
    }

    // ---- online softmax: running max via ds_swizzle xor tree (16-lane halves) ----
#pragma unroll
    for (int r = 0; r < 8; ++r) {
      float mx = __builtin_fmaxf(__builtin_fmaxf(Sacc[0][r], Sacc[1][r]),
                                 __builtin_fmaxf(Sacc[2][r], Sacc[3][r]));
      mx *= inv_scale;
      mx = swz_max_f32<1>(mx);
      mx = swz_max_f32<2>(mx);
      mx = swz_max_f32<4>(mx);
      mx = swz_max_f32<8>(mx);
      const float mnew = __builtin_fmaxf(m_run[r], mx);
      const float a    = __expf(m_run[r] - mnew);
      m_run[r] = mnew;

#pragma unroll
      for (int t = 0; t < 4; ++t) {
        const float p = __expf(Sacc[t][r] * inv_scale - mnew);
        myP[(r + 8 * half) * 64 + 16 * t + lr] = (__bf16)p;   // C-layout scatter
      }
      Lacc[r] *= a;                                  // rescale denominator
#pragma unroll
      for (int t = 0; t < 4; ++t) Oacc[t][r] *= a;   // rescale running output
    }

    // ---- O += P V ; L += P x ones (row sums on the WMMA pipe) ----
#pragma unroll
    for (int kk = 0; kk < 2; ++kk) {
      const __bf16* prow = myP + lr * 64 + kk * 32;
      bf16x16 af;
#pragma unroll
      for (int e = 0; e < 8; ++e) af[e] = prow[16 * half + e];
#pragma unroll
      for (int e = 0; e < 8; ++e) af[8 + e] = prow[16 * half + 8 + e];
      Lacc = WMMA_BF16(af, onesfrag, Lacc);
#pragma unroll
      for (int t = 0; t < 4; ++t) {
        const __bf16* vrow = &Vt[(16 * t + lr) * 64 + kk * 32 + 16 * half];
        bf16x16 bf;
#pragma unroll
        for (int e = 0; e < 16; ++e) bf[e] = vrow[e];
        Oacc[t] = WMMA_BF16(af, bf, Oacc[t]);
      }
    }
    __syncthreads();   // protect Ks/Vt before next iteration's staging
  }

  // ---- finalize: every lane already holds its rows' sums in Lacc ----
#pragma unroll
  for (int t = 0; t < 4; ++t)
#pragma unroll
    for (int r = 0; r < 8; ++r) {
      const float o = Oacc[t][r] / Lacc[r];
      Ob[base + (size_t)(q0 + r + 8 * half) * Dm + 16 * t + lr] = (__bf16)o;
    }
}

// ---------------------------------------------------------------------------
extern "C" void kernel_launch(void* const* d_in, const int* in_sizes, int n_in,
                              void* d_out, int out_size, void* d_ws, size_t ws_size,
                              hipStream_t stream) {
  (void)in_sizes; (void)n_in; (void)out_size; (void)ws_size;
  constexpr int B = 2, S = 4096, D = 512, H = 8;
  constexpr int M = B * S;           // 8192
  constexpr size_t NT = (size_t)B * S * D;

  const float* q  = (const float*)d_in[0];
  const float* k  = (const float*)d_in[1];
  const float* v  = (const float*)d_in[2];
  const float* Wq = (const float*)d_in[3];
  const float* bq = (const float*)d_in[4];
  const float* Wk = (const float*)d_in[5];
  const float* bk = (const float*)d_in[6];
  const float* Wv = (const float*)d_in[7];
  const float* bv = (const float*)d_in[8];
  const float* Wo = (const float*)d_in[9];
  const float* bo = (const float*)d_in[10];

  __bf16* Qb = (__bf16*)d_ws;        // 8 MB each
  __bf16* Kb = Qb + NT;
  __bf16* Vb = Kb + NT;
  __bf16* Ob = Vb + NT;

  dim3 ggrid(M / 128, D / 128);      // (64, 4)
  dim3 gblk(256);

  gemm_bf16_wmma<float, true><<<ggrid, gblk, 0, stream>>>(q, Wq, bq, Qb, M, D, D);
  gemm_bf16_wmma<float, true><<<ggrid, gblk, 0, stream>>>(k, Wk, bk, Kb, M, D, D);
  gemm_bf16_wmma<float, true><<<ggrid, gblk, 0, stream>>>(v, Wv, bv, Vb, M, D, D);

  dim3 agrid(S / 64, H, B);          // (64, 8, 2)
  attn_kernel<<<agrid, dim3(128), 0, stream>>>(Qb, Kb, Vb, Ob);

  gemm_bf16_wmma<__bf16, false><<<ggrid, gblk, 0, stream>>>(Ob, Wo, bo, d_out, M, D, D);
}
